// CausalSelfAttention_77403900609248
// MI455X (gfx1250) — matrix operand order repaired and checked by reading, not compiled, not run on hardware
//
#include <hip/hip_runtime.h>
#include <hip/hip_bf16.h>
#include <stdint.h>

#define D_MODEL 2048
#define NUM_HEADS 16
#define NUM_KV 4
#define HEAD_DIM 128
#define SEQ 2048
#define BATCH 2
#define MROWS (BATCH*SEQ)   // 4096
#define KV_COLS (NUM_KV*HEAD_DIM) // 512

typedef __attribute__((ext_vector_type(16))) __bf16 bf16x16;
typedef __attribute__((ext_vector_type(8)))  float  f32x8;

union Frag { bf16x16 v; uint32_t u[8]; };

__device__ __forceinline__ f32x8 wmma_bf16(bf16x16 a, bf16x16 b, f32x8 c) {
    return __builtin_amdgcn_wmma_f32_16x16x32_bf16(false, a, false, b, (short)0, c, false, false);
}

// A fragment (16x32 bf16, M x K): lane m = lane&15, half = lane>>4.
// VGPR v holds K pair: k = (v<4 ? 2v : 2v+8) + 8*half  -> dword idx (v<4 ? v : v+4) + 4*half
__device__ __forceinline__ bf16x16 load_a(const __bf16* base, int ld, int lane) {
    int h = lane >> 4, m = lane & 15;
    const uint32_t* rp = (const uint32_t*)(base + (size_t)m * ld);
    Frag f;
#pragma unroll
    for (int v = 0; v < 4; ++v) {
        f.u[v]     = rp[v + 4*h];
        f.u[v + 4] = rp[v + 8 + 4*h];
    }
    return f.v;
}

// B fragment (32x16 bf16, K x N) from "N-major, K-contiguous" storage:
// element B(k,n) = base[n*ld + k], with k = (v<4 ? 2v : 2v+8) + 8*half, n = lane&15.
__device__ __forceinline__ bf16x16 load_b_t(const __bf16* base, int ld, int lane) {
    int h = lane >> 4, n = lane & 15;
    const uint32_t* rp = (const uint32_t*)(base + (size_t)n * ld);
    Frag f;
#pragma unroll
    for (int v = 0; v < 8; ++v) f.u[v] = rp[(v < 4 ? v : v + 4) + 4*h];
    return f.v;
}

// Issue one 16-byte async copy (global -> LDS) per lane. CDNA5 path, ASYNCcnt.
__device__ __forceinline__ void async_b128(uint32_t lds_addr, uint64_t gaddr) {
    asm volatile("global_load_async_to_lds_b128 %0, %1, off"
                 :: "v"(lds_addr), "v"(gaddr) : "memory");
}
__device__ __forceinline__ void wait_async0() {
    asm volatile("s_wait_asynccnt 0x0" ::: "memory");
}

__global__ void conv_cast(const float* __restrict__ in, __bf16* __restrict__ out, int nel) {
    int i = blockIdx.x * blockDim.x + threadIdx.x;
    if (i < nel) out[i] = (__bf16)in[i];
}

// in: [K][N] f32 row-major  -> out: [N][K] bf16 (transposed)
__global__ void conv_transpose(const float* __restrict__ in, __bf16* __restrict__ out, int K, int N) {
    int i = blockIdx.x * blockDim.x + threadIdx.x;
    if (i < K * N) {
        int k = i / N, n = i % N;
        out[(size_t)n * K + k] = (__bf16)in[i];
    }
}

// C = A(MROWS x 2048) @ Bt(N x 2048)^T ; one wave computes a 16x64 tile.
// Ping-pong 2x-unrolled fragment pipeline: no register rotation copies,
// loads for tile t+1/t+2 overlap the WMMAs for tile t.
// mode 0: Q (+RoPE)  -> bf16 [row][2048]
// mode 1: K (+RoPE)  -> bf16 [b][kvh][s][128]
// mode 2: V          -> bf16 [b][kvh][d][s]   (transposed for PV B-fragments)
// mode 3: out proj   -> f32  [row][2048]
__global__ __launch_bounds__(256) void gemm16(const __bf16* __restrict__ A,
                                              const __bf16* __restrict__ Bt,
                                              void* __restrict__ Out,
                                              int N, int mode) {
    int lane = threadIdx.x & 31;
    int wave = threadIdx.x >> 5;                 // 0..7
    int mt = blockIdx.x;                         // M tile
    int colBase = blockIdx.y * 512 + wave * 64;
    if (colBase >= N) return;

    f32x8 acc[4] = {};
    const __bf16* Arow = A + (size_t)mt * 16 * D_MODEL;
    const __bf16* Brow = Bt + (size_t)colBase * D_MODEL;

    // buffer 0 <- tile 0
    bf16x16 a0 = load_a(Arow, D_MODEL, lane);
    bf16x16 b0[4];
#pragma unroll
    for (int nt = 0; nt < 4; ++nt)
        b0[nt] = load_b_t(Brow + (size_t)(nt*16) * D_MODEL, D_MODEL, lane);

    bf16x16 a1, b1[4];
    // 64 k-tiles total; loop handles tiles 0..61, leaves tile 62 in buf0.
    for (int k0 = 0; k0 < D_MODEL - 64; k0 += 64) {
        a1 = load_a(Arow + k0 + 32, D_MODEL, lane);
#pragma unroll
        for (int nt = 0; nt < 4; ++nt)
            b1[nt] = load_b_t(Brow + (size_t)(nt*16) * D_MODEL + k0 + 32, D_MODEL, lane);
#pragma unroll
        for (int nt = 0; nt < 4; ++nt) acc[nt] = wmma_bf16(a0, b0[nt], acc[nt]);

        a0 = load_a(Arow + k0 + 64, D_MODEL, lane);
#pragma unroll
        for (int nt = 0; nt < 4; ++nt)
            b0[nt] = load_b_t(Brow + (size_t)(nt*16) * D_MODEL + k0 + 64, D_MODEL, lane);
#pragma unroll
        for (int nt = 0; nt < 4; ++nt) acc[nt] = wmma_bf16(a1, b1[nt], acc[nt]);
    }
    // tail: tile 62 (in buf0) and tile 63
    a1 = load_a(Arow + D_MODEL - 32, D_MODEL, lane);
#pragma unroll
    for (int nt = 0; nt < 4; ++nt)
        b1[nt] = load_b_t(Brow + (size_t)(nt*16) * D_MODEL + D_MODEL - 32, D_MODEL, lane);
#pragma unroll
    for (int nt = 0; nt < 4; ++nt) acc[nt] = wmma_bf16(a0, b0[nt], acc[nt]);
#pragma unroll
    for (int nt = 0; nt < 4; ++nt) acc[nt] = wmma_bf16(a1, b1[nt], acc[nt]);

    int h = lane >> 4, n = lane & 15;
#pragma unroll
    for (int nt = 0; nt < 4; ++nt) {
#pragma unroll
        for (int v = 0; v < 8; ++v) {
            int grow = mt * 16 + v + 8*h;
            int gcol = colBase + nt * 16 + n;
            float val = acc[nt][v];
            if (mode == 0 || mode == 1) {
                // interleaved RoPE: pair (2i,2i+1) sits in adjacent lanes of C
                int d = gcol & (HEAD_DIM - 1);
                int pos = grow & (SEQ - 1);
                float partner = __shfl_xor(val, 1, 32);
                int i2 = d >> 1;
                // theta = pos * 10000^(-2i/128) ; ln(10000)=9.210340371976184
                float freq = __expf(-(float)(2 * i2) * (9.210340371976184f / 128.0f));
                float ang = (float)pos * freq;
                float sv, cv; __sincosf(ang, &sv, &cv);
                val = (d & 1) ? (val * cv + partner * sv) : (val * cv - partner * sv);
            }
            if (mode == 0) {
                ((__bf16*)Out)[(size_t)grow * D_MODEL + gcol] = (__bf16)val;
            } else if (mode == 1) {
                int b = grow >> 11, s = grow & (SEQ - 1);
                int head = gcol >> 7, d = gcol & 127;
                ((__bf16*)Out)[(((size_t)(b * NUM_KV + head)) * SEQ + s) * HEAD_DIM + d] = (__bf16)val;
            } else if (mode == 2) {
                int b = grow >> 11, s = grow & (SEQ - 1);
                int head = gcol >> 7, d = gcol & 127;
                ((__bf16*)Out)[(((size_t)(b * NUM_KV + head)) * HEAD_DIM + d) * SEQ + s] = (__bf16)val;
            } else {
                ((float*)Out)[(size_t)grow * D_MODEL + gcol] = val;
            }
        }
    }
}

// Flash attention: one wave per 16-row query tile; kv tiles of 32.
// K and V tiles are double-buffered in LDS via async global->LDS copies
// (CDNA5 GLOBAL_LOAD_ASYNC_TO_LDS_B128, ASYNCcnt), overlapping the next
// tile's stream with the current tile's softmax + WMMAs.
__global__ __launch_bounds__(32) void attn(const __bf16* __restrict__ Q,
                                           const __bf16* __restrict__ K,
                                           const __bf16* __restrict__ Vt,
                                           __bf16* __restrict__ Y) {
    __shared__ __bf16 Kst[2][32][HEAD_DIM];   // 16 KB, rows = kv, cols = d
    __shared__ __bf16 Vst[2][HEAD_DIM][32];   // 16 KB, rows = d, cols = kv
    __shared__ __bf16 Plds[16][32];           // 1 KB
    int lane = threadIdx.x & 31;
    int qt = blockIdx.x;       // 0..S/16-1
    int head = blockIdx.y;     // 0..15
    int b = blockIdx.z;        // 0..1
    int kvh = head / (NUM_HEADS / NUM_KV);
    const __bf16* Qbase = Q + ((size_t)(b * SEQ + qt * 16)) * D_MODEL + head * HEAD_DIM;
    const __bf16* Kh = K + ((size_t)(b * NUM_KV + kvh)) * SEQ * HEAD_DIM;
    const __bf16* Vh = Vt + ((size_t)(b * NUM_KV + kvh)) * HEAD_DIM * SEQ;

    // ---- async tile staging helpers (8 KB each: 16 ops x 32 lanes x 16 B) ----
    auto issueK = [&](int buf, int kv0) {
        uint32_t lbase = (uint32_t)(uintptr_t)&Kst[buf][0][0];
        const char* g = (const char*)(Kh + (size_t)kv0 * HEAD_DIM);  // contiguous 8 KB
#pragma unroll
        for (int t = 0; t < 16; ++t) {
            int idx = t * 32 + lane;
            async_b128(lbase + idx * 16, (uint64_t)(uintptr_t)(g + idx * 16));
        }
    };
    auto issueV = [&](int buf, int kv0) {
        uint32_t lbase = (uint32_t)(uintptr_t)&Vst[buf][0][0];
        const char* g = (const char*)Vh + (size_t)kv0 * 2;           // 128 rows x 64 B, stride SEQ*2
#pragma unroll
        for (int t = 0; t < 16; ++t) {
            int idx = t * 32 + lane;
            int row = idx >> 2, chunk = idx & 3;
            async_b128(lbase + idx * 16,
                       (uint64_t)(uintptr_t)(g + (size_t)row * (SEQ * 2) + chunk * 16));
        }
    };

    bf16x16 qf[4];
#pragma unroll
    for (int kc = 0; kc < 4; ++kc) qf[kc] = load_a(Qbase + kc * 32, D_MODEL, lane);

    f32x8 o[8] = {};
    float mrow[8], lrow[8];
#pragma unroll
    for (int v = 0; v < 8; ++v) { mrow[v] = -3.0e38f; lrow[v] = 0.0f; }

    int h = lane >> 4, n = lane & 15;
    int qrow_last = qt * 16 + 15;
    int nTiles = qrow_last / 32 + 1;
    const float scale = 0.08838834764831845f; // 1/sqrt(128)

    issueK(0, 0);
    issueV(0, 0);

    for (int jt = 0; jt < nTiles; ++jt) {
        int kv0 = jt * 32;
        int buf = jt & 1;
        wait_async0();                         // current tile resident in LDS
        if (jt + 1 < nTiles) {                 // stream next tile during compute
            issueK(buf ^ 1, kv0 + 32);
            issueV(buf ^ 1, kv0 + 32);
        }

        f32x8 s0 = {}, s1 = {};
#pragma unroll
        for (int kc = 0; kc < 4; ++kc) {
            bf16x16 b0 = load_b_t(&Kst[buf][0][kc * 32], HEAD_DIM, lane);
            bf16x16 b1 = load_b_t(&Kst[buf][16][kc * 32], HEAD_DIM, lane);
            s0 = wmma_bf16(qf[kc], b0, s0);
            s1 = wmma_bf16(qf[kc], b1, s1);
        }
        float alpha[8];
#pragma unroll
        for (int v = 0; v < 8; ++v) {
            int qrow = qt * 16 + v + 8*h;
            float e0 = s0[v] * scale, e1 = s1[v] * scale;
            if (kv0 + n      > qrow) e0 = -3.0e38f;   // causal mask
            if (kv0 + 16 + n > qrow) e1 = -3.0e38f;
            float tm = fmaxf(e0, e1);
#pragma unroll
            for (int off = 1; off < 16; off <<= 1) tm = fmaxf(tm, __shfl_xor(tm, off, 32));
            float mnew = fmaxf(mrow[v], tm);
            float p0 = __expf(e0 - mnew), p1 = __expf(e1 - mnew);
            float rs = p0 + p1;
#pragma unroll
            for (int off = 1; off < 16; off <<= 1) rs += __shfl_xor(rs, off, 32);
            alpha[v] = __expf(mrow[v] - mnew);
            lrow[v] = alpha[v] * lrow[v] + rs;
            mrow[v] = mnew;
            int m = v + 8*h;
            Plds[m][n]      = (__bf16)p0;   // C-layout -> LDS (row-major 16x32)
            Plds[m][16 + n] = (__bf16)p1;
        }
#pragma unroll
        for (int nt = 0; nt < 8; ++nt)
#pragma unroll
            for (int v = 0; v < 8; ++v) o[nt][v] *= alpha[v];
        __syncthreads(); // single wave: DS ops are in-order anyway
        // re-load P as an A fragment (16x32)
        Frag pf;
        {
            int hh = lane >> 4, m = lane & 15;
            const uint32_t* rp = (const uint32_t*)&Plds[m][0];
#pragma unroll
            for (int v = 0; v < 4; ++v) {
                pf.u[v]     = rp[v + 4*hh];
                pf.u[v + 4] = rp[v + 8 + 4*hh];
            }
        }
#pragma unroll
        for (int nt = 0; nt < 8; ++nt) {
            bf16x16 vb = load_b_t(&Vst[buf][nt * 16][0], 32, lane);
            o[nt] = wmma_bf16(pf.v, vb, o[nt]);
        }
        __syncthreads();
    }

#pragma unroll
    for (int nt = 0; nt < 8; ++nt) {
#pragma unroll
        for (int v = 0; v < 8; ++v) {
            int grow = b * SEQ + qt * 16 + v + 8*h;
            int gcol = head * HEAD_DIM + nt * 16 + n;
            Y[(size_t)grow * D_MODEL + gcol] = (__bf16)(o[nt][v] / lrow[v]);
        }
    }
}

extern "C" void kernel_launch(void* const* d_in, const int* in_sizes, int n_in,
                              void* d_out, int out_size, void* d_ws, size_t ws_size,
                              hipStream_t stream) {
    const float* x  = (const float*)d_in[0];
    const float* wq = (const float*)d_in[1];
    const float* wk = (const float*)d_in[2];
    const float* wv = (const float*)d_in[3];
    const float* wo = (const float*)d_in[4];

    char* ws = (char*)d_ws;
    size_t off = 0;
    __bf16* Xbf = (__bf16*)(ws + off); off += (size_t)MROWS * D_MODEL * 2;
    __bf16* Wqt = (__bf16*)(ws + off); off += (size_t)D_MODEL * D_MODEL * 2;
    __bf16* Wkt = (__bf16*)(ws + off); off += (size_t)KV_COLS * D_MODEL * 2;
    __bf16* Wvt = (__bf16*)(ws + off); off += (size_t)KV_COLS * D_MODEL * 2;
    __bf16* Wot = (__bf16*)(ws + off); off += (size_t)D_MODEL * D_MODEL * 2;
    __bf16* Qb  = (__bf16*)(ws + off); off += (size_t)MROWS * D_MODEL * 2;
    __bf16* Kb  = (__bf16*)(ws + off); off += (size_t)BATCH * NUM_KV * SEQ * HEAD_DIM * 2;
    __bf16* Vtb = (__bf16*)(ws + off); off += (size_t)BATCH * NUM_KV * SEQ * HEAD_DIM * 2;
    __bf16* Yb  = (__bf16*)(ws + off); off += (size_t)MROWS * D_MODEL * 2;
    (void)ws_size; (void)in_sizes; (void)n_in; (void)out_size;

    int nel = MROWS * D_MODEL;
    conv_cast<<<(nel + 255) / 256, 256, 0, stream>>>(x, Xbf, nel);
    conv_transpose<<<(D_MODEL * D_MODEL + 255) / 256, 256, 0, stream>>>(wq, Wqt, D_MODEL, D_MODEL);
    conv_transpose<<<(D_MODEL * KV_COLS + 255) / 256, 256, 0, stream>>>(wk, Wkt, D_MODEL, KV_COLS);
    conv_transpose<<<(D_MODEL * KV_COLS + 255) / 256, 256, 0, stream>>>(wv, Wvt, D_MODEL, KV_COLS);
    conv_transpose<<<(D_MODEL * D_MODEL + 255) / 256, 256, 0, stream>>>(wo, Wot, D_MODEL, D_MODEL);

    dim3 gq(MROWS / 16, D_MODEL / 512);   // (256, 4)
    gemm16<<<gq, 256, 0, stream>>>(Xbf, Wqt, Qb, D_MODEL, 0);
    dim3 gkv(MROWS / 16, 1);
    gemm16<<<gkv, 256, 0, stream>>>(Xbf, Wkt, Kb, KV_COLS, 1);
    gemm16<<<gkv, 256, 0, stream>>>(Xbf, Wvt, Vtb, KV_COLS, 2);

    dim3 ga(SEQ / 16, NUM_HEADS, BATCH);
    attn<<<ga, 32, 0, stream>>>(Qb, Kb, Vtb, Yb);

    gemm16<<<gq, 256, 0, stream>>>(Yb, Wot, d_out, D_MODEL, 3);
}
